// GPT_36438502539832
// MI455X (gfx1250) — matrix operand order repaired and checked by reading, not compiled, not run on hardware
//
#include <hip/hip_runtime.h>
#include <cmath>

namespace {

constexpr int L_  = 6;
constexpr int V_  = 512;
constexpr int D_  = 768;
constexpr int P_  = 768;
constexpr int H_  = 12;
constexpr int HD_ = 64;
constexpr int FF_ = 3072;
constexpr int B_  = 8;
constexpr int T_  = 1024;
constexpr int M_  = B_ * T_;

constexpr int PADK_ = 40;   // LDS row stride in bf16 (32 data + 8 pad = 80B) to spread banks

typedef __attribute__((ext_vector_type(16))) __bf16 v16bf;
typedef __attribute__((ext_vector_type(8)))  float  v8f;
typedef __attribute__((ext_vector_type(4)))  unsigned int v4u;
typedef __attribute__((ext_vector_type(8)))  int v8i;
typedef __attribute__((ext_vector_type(4)))  int v4i;

#define HAVE_TDM __has_builtin(__builtin_amdgcn_tensor_load_to_lds)

__device__ __forceinline__ v8f wmma_bf16(v16bf a, v16bf b, v8f c) {
  // D = A(16x32 bf16) * B(32x16 bf16) + C(16x16 f32)
  return __builtin_amdgcn_wmma_f32_16x16x32_bf16(false, a, false, b, (short)0, c,
                                                 false, false);
}

__device__ __forceinline__ v8f zero_v8f() {
  v8f z = {0.f, 0.f, 0.f, 0.f, 0.f, 0.f, 0.f, 0.f};
  return z;
}

// A fragment (16x32, MxK) from row-major [M,K] tile, ld = K stride.
// Lanes 0-15: row = lane, K 0..7 then 16..23 ; lanes 16-31: K 8..15 then 24..31.
__device__ __forceinline__ v16bf load_a_frag(const __bf16* base, int ld) {
  const int lane = threadIdx.x & 31;
  const __bf16* p = base + (size_t)(lane & 15) * ld + ((lane >> 4) << 3);
  v16bf f;
  ((uint4*)&f)[0] = *(const uint4*)(p);
  ((uint4*)&f)[1] = *(const uint4*)(p + 16);
  return f;
}

// B fragment (32x16, KxN) from TRANSPOSED storage Bt[N,K], ld = K stride.
// Lanes 0-15: col N = lane, K 0..7 then 16..23 ; lanes 16-31: col N = lane-16, K 8..15 then 24..31.
__device__ __forceinline__ v16bf load_b_frag(const __bf16* baseT, int ld) {
  const int lane = threadIdx.x & 31;
  const __bf16* p = baseT + (size_t)(lane & 15) * ld + ((lane >> 4) << 3);
  v16bf f;
  ((uint4*)&f)[0] = *(const uint4*)(p);
  ((uint4*)&f)[1] = *(const uint4*)(p + 16);
  return f;
}

// CDNA5 async copy: global -> LDS, 16B per lane, tracked by ASYNCcnt.
__device__ __forceinline__ void async_load_b128(void* lds, const void* gaddr) {
  const unsigned l = (unsigned)(unsigned long long)lds;   // low 32 bits = LDS offset
  asm volatile("global_load_async_to_lds_b128 %0, %1, off"
               :
               : "v"(l), "v"(gaddr)
               : "memory");
}

__device__ __forceinline__ void wait_async0() {
  asm volatile("s_wait_asynccnt 0" ::: "memory");
}

#if HAVE_TDM
// TDM: DMA a 256x32 bf16 tile of row-major tensor [Nrows, K] into LDS,
// inserting 16B of pad after every 32-element row (matches PADK_=40 layout).
// D# encoding per CDNA5 ISA.
__device__ __forceinline__ void tdm_load_b_tile(const __bf16* gsrc, void* lds,
                                                int K, int Nrows) {
  const unsigned long long ga = (unsigned long long)gsrc;
  v4u g0;
  g0[0] = 1u;                                          // count=1, user mode
  g0[1] = (unsigned)(unsigned long long)lds;           // lds_addr (byte)
  g0[2] = (unsigned)ga;                                // global_addr[31:0]
  g0[3] = ((unsigned)(ga >> 32) & 0x1FFFFFFu) | (2u << 30);  // addr[56:32] | type=2
  v8i g1;
  g1[0] = (1 << 16)      // data_size = 2 bytes
        | (1 << 20)      // pad_enable
        | (3 << 22)      // pad_interval: 16 DWORDs (= one 32 x bf16 row)
        | (3 << 25);     // pad_amount:   4 DWORDs (= 16 B)
  g1[1] = (K & 0xFFFF) << 16;                          // tensor_dim0[15:0]
  g1[2] = ((K >> 16) & 0xFFFF) | ((Nrows & 0xFFFF) << 16);   // td0 hi | td1 lo
  g1[3] = ((Nrows >> 16) & 0xFFFF) | (32 << 16);       // td1 hi | tile_dim0 = 32
  g1[4] = 256;                                         // tile_dim1 = 256, tile_dim2 = 0
  g1[5] = K;                                           // tensor_dim0_stride[31:0]
  g1[6] = 0;                                           // t0s hi | t1s lo
  g1[7] = 0;
  const v4i z4 = {0, 0, 0, 0};                         // 2-D tensor: groups 2/3 unused
  const v8i z8 = {0, 0, 0, 0, 0, 0, 0, 0};
  __builtin_amdgcn_tensor_load_to_lds(g0, g1, z4, z4, z8, 0);
}
#endif

// ---------------- embedding gather: x = tok_emb[idx] + pos_emb ----------------
__global__ __launch_bounds__(256) void embed_kernel(const int* __restrict__ idx,
                                                    const float* __restrict__ tok,
                                                    const float* __restrict__ pos,
                                                    float* __restrict__ x) {
  const int row = blockIdx.x;            // 0..M-1
  const int t   = row & (T_ - 1);
  const int tk  = idx[row];
  #pragma unroll
  for (int j = 0; j < 3; ++j) {
    const int c = threadIdx.x + j * 256;
    x[(size_t)row * D_ + c] = tok[(size_t)tk * D_ + c] + pos[(size_t)t * D_ + c];
  }
}

// ---------------- layernorm (fp32 in) -> bf16 out ----------------
__global__ __launch_bounds__(256) void ln_kernel(const float* __restrict__ x,
                                                 const float* __restrict__ g,
                                                 const float* __restrict__ bta,
                                                 __bf16* __restrict__ out) {
  __shared__ float red[256];
  const int row = blockIdx.x;
  const int tid = threadIdx.x;
  const float* xr = x + (size_t)row * D_;
  float v0 = xr[tid], v1 = xr[tid + 256], v2 = xr[tid + 512];
  red[tid] = v0 + v1 + v2;
  __syncthreads();
  for (int off = 128; off > 0; off >>= 1) {
    if (tid < off) red[tid] += red[tid + off];
    __syncthreads();
  }
  const float mu = red[0] * (1.0f / D_);
  __syncthreads();
  const float d0 = v0 - mu, d1 = v1 - mu, d2 = v2 - mu;
  red[tid] = d0 * d0 + d1 * d1 + d2 * d2;
  __syncthreads();
  for (int off = 128; off > 0; off >>= 1) {
    if (tid < off) red[tid] += red[tid + off];
    __syncthreads();
  }
  const float rstd = rsqrtf(red[0] * (1.0f / D_) + 1e-5f);
  __bf16* orow = out + (size_t)row * D_;
  orow[tid]       = (__bf16)(d0 * rstd * g[tid]       + bta[tid]);
  orow[tid + 256] = (__bf16)(d1 * rstd * g[tid + 256] + bta[tid + 256]);
  orow[tid + 512] = (__bf16)(d2 * rstd * g[tid + 512] + bta[tid + 512]);
}

// -------- fp32 [K,N] -> bf16 transposed [N,K] (weights into WMMA-B layout) ----
__global__ __launch_bounds__(256) void convt_kernel(const float* __restrict__ W,
                                                    __bf16* __restrict__ Wt,
                                                    int K, int N) {
  const int tid = blockIdx.x * 256 + threadIdx.x;
  if (tid >= K * N) return;
  const int k = tid / N, n = tid % N;
  Wt[(size_t)n * K + k] = (__bf16)W[tid];
}

// -------- V [B*T, H*HD] bf16 -> Vt [B,H,HD,T] bf16 ----------------------------
__global__ __launch_bounds__(256) void vtrans_kernel(const __bf16* __restrict__ v,
                                                     __bf16* __restrict__ vT) {
  const int tid = blockIdx.x * 256 + threadIdx.x;   // < M_*P_
  const int row = tid / P_, col = tid % P_;
  const int b = row / T_, t = row % T_;
  const int h = col / HD_, hd = col % HD_;
  vT[(((size_t)(b * H_ + h)) * HD_ + hd) * T_ + t] = v[tid];
}

// ---------------- WMMA GEMM: C[M,N] = A[M,K](bf16) * Bt[N,K](bf16) ------------
// Block = 256 thr (8 waves) -> 64x256 of C; wave -> 32x64 (2x4 WMMA tiles).
// B tile (256x32) staged by the Tensor Data Mover (TENSORcnt); A tile (64x32)
// by per-lane async global->LDS copies (ASYNCcnt). Double-buffered.
template <bool BIAS, bool GACT, bool RESID, bool OUTB>
__global__ __launch_bounds__(256) void gemm_bf16(const __bf16* __restrict__ A,
                                                 const __bf16* __restrict__ Bt,
                                                 const float* __restrict__ bias,
                                                 const float* __restrict__ resid,
                                                 float* __restrict__ Cf,
                                                 __bf16* __restrict__ Cb,
                                                 int M, int N, int K) {
  __shared__ __align__(16) __bf16 lds_a[2][64 * PADK_];
  __shared__ __align__(16) __bf16 lds_b[2][256 * PADK_];
  const int tid  = threadIdx.x;
  const int wid  = tid >> 5;
  const int lane = tid & 31;
  const int r    = lane & 15;
  const int hi   = lane >> 4;
  const int mblk = blockIdx.y * 64;
  const int nblk = blockIdx.x * 256;
  const int am0  = (wid >> 2) * 32;    // wave's local row base within the block tile
  const int bn0  = (wid & 3) * 64;     // wave's local col base within the block tile
  (void)M;

  const int arow = tid >> 2, asub = tid & 3;   // 64 rows x 4 16B-chunks of the A tile

  auto issue = [&](int kk, int buf) {
    // A tile: 64x32 bf16 (256 x 16B chunks; one per thread) via async copy
    async_load_b128(&lds_a[buf][arow * PADK_ + asub * 8],
                    A + (size_t)(mblk + arow) * K + kk + asub * 8);
#if HAVE_TDM
    // B tile: 256x32 bf16, one TDM descriptor issued by wave 0
    if (wid == 0)
      tdm_load_b_tile(Bt + (size_t)nblk * K + kk, &lds_b[buf][0], K, N);
#else
    #pragma unroll
    for (int j = 0; j < 4; ++j) {
      const int c = tid + 256 * j;
      const int row = c >> 2, sub = c & 3;
      async_load_b128(&lds_b[buf][row * PADK_ + sub * 8],
                      Bt + (size_t)(nblk + row) * K + kk + sub * 8);
    }
#endif
  };

  v8f c[2][4];
  #pragma unroll
  for (int i = 0; i < 2; ++i)
    #pragma unroll
    for (int j = 0; j < 4; ++j) c[i][j] = zero_v8f();

  issue(0, 0);
  const int steps = K / 32;
  int cur = 0;
  for (int s = 0; s < steps; ++s) {
    wait_async0();                                  // my A-tile copies landed
#if HAVE_TDM
    __builtin_amdgcn_s_wait_tensorcnt((short)0);    // wave 0: B-tile DMA landed
#endif
    __syncthreads();                                // whole block: buffer `cur` ready
    if (s + 1 < steps) issue((s + 1) * 32, cur ^ 1);   // prefetch next K-slab

    v16bf a0 = load_a_frag(&lds_a[cur][(size_t)am0 * PADK_], PADK_);
    v16bf a1 = load_a_frag(&lds_a[cur][(size_t)(am0 + 16) * PADK_], PADK_);
    #pragma unroll
    for (int j = 0; j < 4; ++j) {
      v16bf bf = load_b_frag(&lds_b[cur][(size_t)(bn0 + 16 * j) * PADK_], PADK_);
      c[0][j] = wmma_bf16(a0, bf, c[0][j]);
      c[1][j] = wmma_bf16(a1, bf, c[1][j]);
    }
    cur ^= 1;
  }

  #pragma unroll
  for (int i = 0; i < 2; ++i) {
    #pragma unroll
    for (int j = 0; j < 4; ++j) {
      const int col = nblk + bn0 + 16 * j + r;
      const float bv = BIAS ? bias[col] : 0.0f;
      #pragma unroll
      for (int e = 0; e < 8; ++e) {
        const int row = mblk + am0 + 16 * i + e + 8 * hi;
        float val = c[i][j][e] + bv;
        if (GACT)  val = 0.5f * val * (1.0f + erff(val * 0.70710678118654752f));
        if (RESID) val += resid[(size_t)row * N + col];
        if (OUTB)  Cb[(size_t)row * N + col] = (__bf16)val;
        else       Cf[(size_t)row * N + col] = val;
      }
    }
  }
}

// ---------------- flash attention: wave handles a 16-row query tile -----------
__global__ __launch_bounds__(256) void attn_fwd(const __bf16* __restrict__ q,
                                                const __bf16* __restrict__ kbuf,
                                                const __bf16* __restrict__ vT,
                                                const float* __restrict__ amask,
                                                __bf16* __restrict__ y) {
  __shared__ __align__(16) __bf16 lds_p[8][16 * 32];   // per-wave P tile (16x32)
  const int bh   = blockIdx.y;
  const int b    = bh / H_;
  const int h    = bh % H_;
  const int wid  = threadIdx.x >> 5;
  const int lane = threadIdx.x & 31;
  const int r    = lane & 15;
  const int hi   = lane >> 4;
  const int q0   = blockIdx.x * 128 + wid * 16;

  const __bf16* qbase = q + ((size_t)(b * T_ + q0)) * P_ + h * HD_;
  const v16bf aq0 = load_a_frag(qbase, P_);        // kd 0..31
  const v16bf aq1 = load_a_frag(qbase + 32, P_);   // kd 32..63

  v8f o0 = zero_v8f(), o1 = zero_v8f(), o2 = zero_v8f(), o3 = zero_v8f();
  float m_i[8], l_i[8];
  #pragma unroll
  for (int e = 0; e < 8; ++e) { m_i[e] = -1e30f; l_i[e] = 0.0f; }

  const int nch = (q0 + 15) / 32 + 1;   // causal frontier in 32-wide key chunks
  for (int ch = 0; ch < nch; ++ch) {
    const int k0 = ch * 32;
    const __bf16* kbase = kbuf + ((size_t)(b * T_ + k0)) * P_ + h * HD_;
    v8f s0 = zero_v8f(), s1 = zero_v8f();
    {
      v16bf b00 = load_b_frag(kbase, P_);                        // tk k0..+15, kd 0..31
      v16bf b01 = load_b_frag(kbase + 32, P_);                   // kd 32..63
      s0 = wmma_bf16(aq0, b00, s0);
      s0 = wmma_bf16(aq1, b01, s0);
      v16bf b10 = load_b_frag(kbase + (size_t)16 * P_, P_);      // tk k0+16..+31
      v16bf b11 = load_b_frag(kbase + (size_t)16 * P_ + 32, P_);
      s1 = wmma_bf16(aq0, b10, s1);
      s1 = wmma_bf16(aq1, b11, s1);
    }
    const int c0 = k0 + r;
    const int c1 = k0 + 16 + r;
    const float am0v = amask[b * T_ + c0];
    const float am1v = amask[b * T_ + c1];
    #pragma unroll
    for (int e = 0; e < 8; ++e) {
      const int row = q0 + e + 8 * hi;
      float v0 = s0[e] * 0.125f;                    // 1/sqrt(64)
      float v1 = s1[e] * 0.125f;
      if (c0 > row || am0v == 0.0f) v0 = -1e30f;
      if (c1 > row || am1v == 0.0f) v1 = -1e30f;
      float rm = fmaxf(v0, v1);                     // row max over 16-lane group
      rm = fmaxf(rm, __shfl_xor(rm, 1, 32));
      rm = fmaxf(rm, __shfl_xor(rm, 2, 32));
      rm = fmaxf(rm, __shfl_xor(rm, 4, 32));
      rm = fmaxf(rm, __shfl_xor(rm, 8, 32));
      const float mnew  = fmaxf(m_i[e], rm);
      const float alpha = __expf(m_i[e] - mnew);
      m_i[e] = mnew;
      const float p0 = __expf(v0 - mnew);
      const float p1 = __expf(v1 - mnew);
      float ps = p0 + p1;
      ps += __shfl_xor(ps, 1, 32);
      ps += __shfl_xor(ps, 2, 32);
      ps += __shfl_xor(ps, 4, 32);
      ps += __shfl_xor(ps, 8, 32);
      l_i[e] = l_i[e] * alpha + ps;
      o0[e] *= alpha; o1[e] *= alpha; o2[e] *= alpha; o3[e] *= alpha;
      // re-layout P (C/D striping) -> row-major 16x32 in LDS for A-frag reload
      lds_p[wid][(e + 8 * hi) * 32 + r]      = (__bf16)p0;
      lds_p[wid][(e + 8 * hi) * 32 + 16 + r] = (__bf16)p1;
    }
    asm volatile("s_wait_dscnt 0" ::: "memory");
    const v16bf pf = load_a_frag(&lds_p[wid][0], 32);
    const __bf16* vbase = vT + (((size_t)(b * H_ + h)) * HD_) * T_ + k0;
    o0 = wmma_bf16(pf, load_b_frag(vbase + (size_t)0  * T_, T_), o0);
    o1 = wmma_bf16(pf, load_b_frag(vbase + (size_t)16 * T_, T_), o1);
    o2 = wmma_bf16(pf, load_b_frag(vbase + (size_t)32 * T_, T_), o2);
    o3 = wmma_bf16(pf, load_b_frag(vbase + (size_t)48 * T_, T_), o3);
  }

  #pragma unroll
  for (int e = 0; e < 8; ++e) {
    const float inv = 1.0f / l_i[e];
    const int row = q0 + e + 8 * hi;
    __bf16* yr = y + ((size_t)(b * T_ + row)) * P_ + h * HD_ + r;
    yr[0]  = (__bf16)(o0[e] * inv);
    yr[16] = (__bf16)(o1[e] * inv);
    yr[32] = (__bf16)(o2[e] * inv);
    yr[48] = (__bf16)(o3[e] * inv);
  }
}

}  // namespace

extern "C" void kernel_launch(void* const* d_in, const int* in_sizes, int n_in,
                              void* d_out, int out_size, void* d_ws, size_t ws_size,
                              hipStream_t stream) {
  (void)in_sizes; (void)n_in; (void)out_size; (void)ws_size;
  const int*   idx   = (const int*)d_in[0];
  const float* amask = (const float*)d_in[1];
  const float* tok   = (const float*)d_in[2];
  const float* pos   = (const float*)d_in[3];
  const float* ln1g  = (const float*)d_in[4];
  const float* ln1b  = (const float*)d_in[5];
  const float* Wq    = (const float*)d_in[6];
  const float* bq    = (const float*)d_in[7];
  const float* Wk    = (const float*)d_in[8];
  const float* bk    = (const float*)d_in[9];
  const float* Wv    = (const float*)d_in[10];
  const float* bv    = (const float*)d_in[11];
  const float* Wo    = (const float*)d_in[12];
  const float* bo    = (const float*)d_in[13];
  const float* ln2g  = (const float*)d_in[14];
  const float* ln2b  = (const float*)d_in[15];
  const float* W1    = (const float*)d_in[16];
  const float* b1    = (const float*)d_in[17];
  const float* W2    = (const float*)d_in[18];
  const float* b2    = (const float*)d_in[19];
  const float* lnfg  = (const float*)d_in[20];
  const float* lnfb  = (const float*)d_in[21];
  const float* Wout  = (const float*)d_in[22];

  char* ws = (char*)d_ws;
  size_t off = 0;
  auto carve = [&](size_t bytes) -> void* {
    void* p = ws + off;
    off = (off + bytes + 255) & ~(size_t)255;
    return p;
  };
  float*  x    = (float*) carve((size_t)M_ * D_ * 4);
  __bf16* hb   = (__bf16*)carve((size_t)M_ * D_ * 2);
  __bf16* qb   = (__bf16*)carve((size_t)M_ * P_ * 2);
  __bf16* kb   = (__bf16*)carve((size_t)M_ * P_ * 2);
  __bf16* vb   = (__bf16*)carve((size_t)M_ * P_ * 2);
  __bf16* vTb  = (__bf16*)carve((size_t)M_ * P_ * 2);
  __bf16* yb   = (__bf16*)carve((size_t)M_ * P_ * 2);
  __bf16* m1b  = (__bf16*)carve((size_t)M_ * FF_ * 2);
  __bf16* wqt  = (__bf16*)carve((size_t)D_ * P_ * 2);
  __bf16* wkt  = (__bf16*)carve((size_t)D_ * P_ * 2);
  __bf16* wvt  = (__bf16*)carve((size_t)D_ * P_ * 2);
  __bf16* wot  = (__bf16*)carve((size_t)P_ * D_ * 2);
  __bf16* w1t  = (__bf16*)carve((size_t)D_ * FF_ * 2);
  __bf16* w2t  = (__bf16*)carve((size_t)FF_ * D_ * 2);
  __bf16* woutt= (__bf16*)carve((size_t)D_ * V_ * 2);

  const dim3 blk(256);
  const int nDP = D_ * P_;
  const int nDF = D_ * FF_;

  embed_kernel<<<M_, blk, 0, stream>>>(idx, tok, pos, x);

  for (int l = 0; l < L_; ++l) {
    ln_kernel<<<M_, blk, 0, stream>>>(x, ln1g + l * D_, ln1b + l * D_, hb);

    convt_kernel<<<(nDP + 255) / 256, blk, 0, stream>>>(Wq + (size_t)l * nDP, wqt, D_, P_);
    convt_kernel<<<(nDP + 255) / 256, blk, 0, stream>>>(Wk + (size_t)l * nDP, wkt, D_, P_);
    convt_kernel<<<(nDP + 255) / 256, blk, 0, stream>>>(Wv + (size_t)l * nDP, wvt, D_, P_);
    convt_kernel<<<(nDP + 255) / 256, blk, 0, stream>>>(Wo + (size_t)l * nDP, wot, P_, D_);
    convt_kernel<<<(nDF + 255) / 256, blk, 0, stream>>>(W1 + (size_t)l * nDF, w1t, D_, FF_);
    convt_kernel<<<(nDF + 255) / 256, blk, 0, stream>>>(W2 + (size_t)l * nDF, w2t, FF_, D_);

    const dim3 gqkv(P_ / 256, M_ / 64);
    gemm_bf16<true, false, false, true><<<gqkv, blk, 0, stream>>>(
        hb, wqt, bq + (size_t)l * P_, nullptr, nullptr, qb, M_, P_, D_);
    gemm_bf16<true, false, false, true><<<gqkv, blk, 0, stream>>>(
        hb, wkt, bk + (size_t)l * P_, nullptr, nullptr, kb, M_, P_, D_);
    gemm_bf16<true, false, false, true><<<gqkv, blk, 0, stream>>>(
        hb, wvt, bv + (size_t)l * P_, nullptr, nullptr, vb, M_, P_, D_);

    vtrans_kernel<<<(M_ * P_ + 255) / 256, blk, 0, stream>>>(vb, vTb);

    attn_fwd<<<dim3(T_ / 128, B_ * H_), blk, 0, stream>>>(qb, kb, vTb, amask, yb);

    gemm_bf16<true, false, true, false><<<dim3(D_ / 256, M_ / 64), blk, 0, stream>>>(
        yb, wot, bo + (size_t)l * D_, x, x, nullptr, M_, D_, P_);

    ln_kernel<<<M_, blk, 0, stream>>>(x, ln2g + l * D_, ln2b + l * D_, hb);

    gemm_bf16<true, true, false, true><<<dim3(FF_ / 256, M_ / 64), blk, 0, stream>>>(
        hb, w1t, b1 + (size_t)l * FF_, nullptr, nullptr, m1b, M_, FF_, D_);
    gemm_bf16<true, false, true, false><<<dim3(D_ / 256, M_ / 64), blk, 0, stream>>>(
        m1b, w2t, b2 + (size_t)l * D_, x, x, nullptr, M_, D_, FF_);
  }

  ln_kernel<<<M_, blk, 0, stream>>>(x, lnfg, lnfb, hb);
  convt_kernel<<<(D_ * V_ + 255) / 256, blk, 0, stream>>>(Wout, woutt, D_, V_);
  gemm_bf16<false, false, false, false><<<dim3(V_ / 256, M_ / 64), blk, 0, stream>>>(
      hb, woutt, nullptr, nullptr, (float*)d_out, nullptr, M_, V_, D_);
}
